// Decoder_3109556322707
// MI455X (gfx1250) — matrix operand order repaired and checked by reading, not compiled, not run on hardware
//
#include <hip/hip_runtime.h>

// ---------------------------------------------------------------------------
// Decoder step for MI455X (gfx1250, wave32).
// Dominant work: fused 5-way conv1x1 == GEMM M=512, N=32*1000, K=5*512=2560
// done with v_wmma_f32_16x16x32_f16 (f16 in, f32 accumulate), tanh + v-
// reduction fused in registers. A-tiles staged with CDNA5 async global->LDS
// (ASYNCcnt), double-buffered LDS pipeline; X streamed with NT hints.
// ---------------------------------------------------------------------------

#define Hdim  512
#define Bdim  32
#define Ndim  1000
#define KTOT  2560          // 5*H
#define NTILE 128
#define KC    32
#define KSTEPS (KTOT / KC)  // 80
#define NCOLS (Bdim * Ndim) // 32000 columns, 250 tiles of 128

typedef __attribute__((ext_vector_type(16))) _Float16 v16h;
typedef __attribute__((ext_vector_type(8)))  _Float16 v8h;
typedef __attribute__((ext_vector_type(8)))  float    v8f;
typedef __attribute__((ext_vector_type(4)))  float    v4f;

union H16 { v16h v; v8h h[2]; };
union PK2 { _Float16 h[2]; uint32_t u; };

__device__ __forceinline__ v8f wmma_16x16x32(v16h a, v16h b, v8f c) {
  // 8 args: (neg_a, A, neg_b, B, c_mod, C, reuse_a, reuse_b)
  return __builtin_amdgcn_wmma_f32_16x16x32_f16(false, a, false, b,
                                                (short)0, c, false, false);
}

// CDNA5 async copy: 64B row = 4x global_load_async_to_lds_b128, same base
// VGPRs, INST_OFFSET applies to both global and LDS sides (ISA 08 §4.4).
__device__ __forceinline__ void async_row64(void* lds_dst, const void* gsrc) {
  uint32_t lo = (uint32_t)(uintptr_t)lds_dst;   // low 32 bits == LDS offset
  uint64_t ga = (uint64_t)(uintptr_t)gsrc;
  asm volatile(
      "global_load_async_to_lds_b128 %0, %1, off\n\t"
      "global_load_async_to_lds_b128 %0, %1, off offset:16\n\t"
      "global_load_async_to_lds_b128 %0, %1, off offset:32\n\t"
      "global_load_async_to_lds_b128 %0, %1, off offset:48"
      :: "v"(lo), "v"(ga) : "memory");
}
__device__ __forceinline__ void wait_asynccnt0() {
  asm volatile("s_wait_asynccnt 0x0" ::: "memory");
}

__device__ __forceinline__ float sigm(float x) {
  return 1.0f / (1.0f + __expf(-x));
}

// ---------------------------------------------------------------------------
// k_prep_w: pack [Wref|Wd|Wdr|Wdist|Wchl] into row-major f16 Wcat[512][2560]
// ---------------------------------------------------------------------------
__global__ void k_prep_w(const float* __restrict__ Wref, const float* __restrict__ Wd,
                         const float* __restrict__ Wdr,  const float* __restrict__ Wdist,
                         const float* __restrict__ Wchl, _Float16* __restrict__ Wf16) {
  int idx = blockIdx.x * 256 + threadIdx.x;        // 0 .. 512*2560-1
  int m   = idx / KTOT;
  int col = idx - m * KTOT;
  int s   = col >> 9;
  int k   = col & 511;
  const float* src = (s == 0) ? Wref : (s == 1) ? Wd : (s == 2) ? Wdr
                   : (s == 3) ? Wdist : Wchl;
  Wf16[(size_t)m * KTOT + col] = (_Float16)src[m * Hdim + k];
}

// ---------------------------------------------------------------------------
// k_gates: gates(B,4H) = x @ W_ih.T + h0 @ W_hh.T   (tiny; VALU f32)
// ---------------------------------------------------------------------------
__global__ void k_gates(const float* __restrict__ dec_in, const float* __restrict__ h0,
                        const float* __restrict__ Wih,    const float* __restrict__ Whh,
                        float* __restrict__ gates) {
  __shared__ float xs[8 * Hdim];
  __shared__ float hs[8 * Hdim];
  int tid = threadIdx.x;
  int b0  = blockIdx.x * 8;
  for (int idx = tid; idx < 8 * Hdim; idx += 256) {
    int b = idx >> 9, k = idx & 511;
    xs[idx] = dec_in[(size_t)(b0 + b) * Hdim + k];  // decoder_input[:,:,0]
    hs[idx] = h0[(size_t)(b0 + b) * Hdim + k];
  }
  __syncthreads();
  for (int oo = 0; oo < 8; ++oo) {
    int o = oo * 256 + tid;                         // 0..2047
    float acc[8];
#pragma unroll
    for (int b = 0; b < 8; ++b) acc[b] = 0.f;
    for (int k = 0; k < Hdim; ++k) {
      float w1 = Wih[(size_t)o * Hdim + k];
      float w2 = Whh[(size_t)o * Hdim + k];
#pragma unroll
      for (int b = 0; b < 8; ++b)
        acc[b] += w1 * xs[b * Hdim + k] + w2 * hs[b * Hdim + k];
    }
    for (int b = 0; b < 8; ++b)
      gates[(size_t)(b0 + b) * (4 * Hdim) + o] = acc[b];
  }
}

// ---------------------------------------------------------------------------
// k_cell: LSTM cell pointwise; writes h_new, c_new into d_out tail + ws copy
// ---------------------------------------------------------------------------
__global__ void k_cell(const float* __restrict__ gates, const float* __restrict__ c0,
                       float* __restrict__ out, float* __restrict__ hcopy) {
  int idx = blockIdx.x * 256 + threadIdx.x;   // b*H + h
  int b = idx >> 9, h = idx & 511;
  const float* g = gates + (size_t)b * (4 * Hdim);
  float ii = g[h], ff = g[Hdim + h], gg = g[2 * Hdim + h], oo = g[3 * Hdim + h];
  float cn = sigm(ff) * c0[idx] + sigm(ii) * tanhf(gg);
  float hn = sigm(oo) * tanhf(cn);
  out[32000 + idx] = hn;                      // h_new
  out[32000 + Bdim * Hdim + idx] = cn;        // c_new
  hcopy[idx] = hn;
}

// ---------------------------------------------------------------------------
// k_qplus: qplus(B,H) = h_new @ Wq.T + bq + (bd+bdr+bdist+bchl+bref)
// ---------------------------------------------------------------------------
__global__ void k_qplus(const float* __restrict__ hcopy, const float* __restrict__ Wq,
                        const float* __restrict__ bq,  const float* __restrict__ bd,
                        const float* __restrict__ bdr, const float* __restrict__ bdist,
                        const float* __restrict__ bchl,const float* __restrict__ bref,
                        float* __restrict__ qplus) {
  int idx = blockIdx.x * 256 + threadIdx.x;   // b*H + h
  int b = idx >> 9, h = idx & 511;
  float acc = bq[h] + bd[h] + bdr[h] + bdist[h] + bchl[h] + bref[h];
  const float* hp = hcopy + (size_t)b * Hdim;
  const float* wp = Wq + (size_t)h * Hdim;
  for (int k = 0; k < Hdim; ++k) acc += wp[k] * hp[k];
  qplus[idx] = acc;
}

// ---------------------------------------------------------------------------
// k_main: GEMM(512 x 32000, K=2560) via WMMA + fused tanh + v-reduction.
// grid 250 x 512 threads (16 waves). Wave (mgrp,ngrp): M range mgrp*128..+127,
// N range ngrp*32..+31 -> 8x2 = 16 accum tiles of 16x16 (v8f each).
// Double-buffered LDS; A staged with async global->LDS b128; X loads NT.
// ---------------------------------------------------------------------------
__global__ __launch_bounds__(512, 1)
void k_main(const _Float16* __restrict__ Wf16,
            const float* __restrict__ X0, const float* __restrict__ X1,
            const float* __restrict__ X2, const float* __restrict__ X3,
            const float* __restrict__ X4,
            const float* __restrict__ qplus, const float* __restrict__ v,
            float* __restrict__ out) {
  __shared__ __align__(16) _Float16 lA[2][512 * KC];   // [m][k], 64B rows, 2x32KB
  __shared__ __align__(16) _Float16 lB[2][NTILE * 40]; // [c][k], 80B rows, 2x10KB
  __shared__ float lqb[2 * Hdim];
  __shared__ float lv[Hdim];
  __shared__ float llog[NTILE];

  const int tid  = threadIdx.x;
  const int lane = tid & 31;
  const int wave = tid >> 5;
  const int grp  = lane >> 4;        // half-wave select
  const int l15  = lane & 15;
  const int mgrp = wave & 3;         // M base = mgrp*128
  const int ngrp = wave >> 2;        // N base = ngrp*32

  const int c0   = blockIdx.x * NTILE;       // flattened (b,n) column base
  const int b_lo = c0 / Ndim;
  const int thr  = (b_lo + 1) * Ndim - c0;   // local col >= thr -> batch b_lo+1
  const int b_hi = (b_lo < Bdim - 1) ? b_lo + 1 : b_lo;

  // stage per-column bias rows (q + summed conv biases), v, zero logits
  for (int idx = tid; idx < 2 * Hdim; idx += 512) {
    int which = idx >> 9, m = idx & 511;
    lqb[idx] = qplus[(size_t)(which ? b_hi : b_lo) * Hdim + m];
  }
  lv[tid] = v[tid & 511];
  if (tid < NTILE) llog[tid] = 0.f;

  // B-stage thread mapping: (k,k+1) pair x 4 columns per thread.
  // thr is 4-aligned (1000 % 4 == 0, c0 % 4 == 0) so a float4 never straddles b.
  const int kp = tid >> 5;                // 0..15 -> k pair base 2*kp
  const int cg = (tid & 31) * 4;          // column group base 0..124

  v8f acc[8][2];
  {
    v8f z = {0.f, 0.f, 0.f, 0.f, 0.f, 0.f, 0.f, 0.f};
#pragma unroll
    for (int mt = 0; mt < 8; ++mt) { acc[mt][0] = z; acc[mt][1] = z; }
  }

  // ---- staging helpers ----
  auto stageA = [&](int ks, int buf) {
    async_row64(&lA[buf][tid * KC], Wf16 + (size_t)tid * KTOT + ks * KC);
  };
  auto stageB = [&](int ks, int buf) {
    int kg  = ks * KC + kp * 2;       // even; kg,kg+1 in same 512-segment
    int seg = kg >> 9;
    int kk  = kg & 511;
    const float* xp = (seg == 0) ? X0 : (seg == 1) ? X1 : (seg == 2) ? X2
                    : (seg == 3) ? X3 : X4;
    int b = (cg >= thr) ? b_lo + 1 : b_lo;
    int n = c0 + cg - b * Ndim;
    const v4f* p0 = (const v4f*)(xp + ((size_t)b * Hdim + kk) * Ndim + n);
    const v4f* p1 = (const v4f*)(xp + ((size_t)b * Hdim + kk + 1) * Ndim + n);
    v4f r0 = __builtin_nontemporal_load(p0);   // stream-once: NT hint
    v4f r1 = __builtin_nontemporal_load(p1);
    _Float16* dst = &lB[buf][0];
#pragma unroll
    for (int i = 0; i < 4; ++i) {
      PK2 pk;
      pk.h[0] = (_Float16)r0[i];
      pk.h[1] = (_Float16)r1[i];
      *(uint32_t*)&dst[(cg + i) * 40 + kp * 2] = pk.u;   // 4B-aligned b32 store
    }
  };

  // ---- prologue: fill buffer 0 ----
  stageA(0, 0);
  stageB(0, 0);
  wait_asynccnt0();
  __syncthreads();

  for (int ks = 0; ks < KSTEPS; ++ks) {
    const int cur = ks & 1;
    // issue next-step staging into the other buffer (overlaps with compute)
    if (ks + 1 < KSTEPS) {
      stageA(ks + 1, cur ^ 1);
      stageB(ks + 1, cur ^ 1);
    }
    // ---- WMMA on current buffer ----
    // B frags per-lane k {0-7,16-23 | 8-15,24-31}; A per-lane k {0-7,16-23 | 8-15,24-31}
    H16 bf0, bf1;
    {
      const _Float16* cp = &lB[cur][(ngrp * 32 + l15) * 40 + grp * 8];
      bf0.h[0] = *(const v8h*)cp;
      bf0.h[1] = *(const v8h*)(cp + 16);
      const _Float16* cp1 = cp + 16 * 40;
      bf1.h[0] = *(const v8h*)cp1;
      bf1.h[1] = *(const v8h*)(cp1 + 16);
    }
#pragma unroll
    for (int mt = 0; mt < 8; ++mt) {
      const _Float16* rp = &lA[cur][(mgrp * 128 + mt * 16 + l15) * KC + grp * 8];
      H16 af;
      af.h[0] = *(const v8h*)rp;
      af.h[1] = *(const v8h*)(rp + 16);
      acc[mt][0] = wmma_16x16x32(af.v, bf0.v, acc[mt][0]);
      acc[mt][1] = wmma_16x16x32(af.v, bf1.v, acc[mt][1]);
    }
    // async A writes must land before other waves read past the barrier
    wait_asynccnt0();
    __syncthreads();
  }

  // ---- epilogue: tanh(out + qplus[b(col)]) * v[m], reduce over m ----
  {
    float a0 = 0.f, a1 = 0.f;
    const int cl0  = ngrp * 32 + l15;            // nt=0 column, nt=1 is +16
    const int sel0 = (cl0 >= thr) ? Hdim : 0;
    const int sel1 = (cl0 + 16 >= thr) ? Hdim : 0;
#pragma unroll
    for (int mt = 0; mt < 8; ++mt) {
#pragma unroll
      for (int j = 0; j < 8; ++j) {
        int m = mgrp * 128 + mt * 16 + grp * 8 + j;  // C VGPR j -> row j / j+8
        float vm = lv[m];
        a0 += vm * tanhf(acc[mt][0][j] + lqb[sel0 + m]);
        a1 += vm * tanhf(acc[mt][1][j] + lqb[sel1 + m]);
      }
    }
    atomicAdd(&llog[cl0], a0);
    atomicAdd(&llog[cl0 + 16], a1);
  }
  __syncthreads();
  if (tid < NTILE) out[c0 + tid] = llog[tid];
}

// ---------------------------------------------------------------------------
extern "C" void kernel_launch(void* const* d_in, const int* in_sizes, int n_in,
                              void* d_out, int out_size, void* d_ws, size_t ws_size,
                              hipStream_t stream) {
  const float* static_hidden = (const float*)d_in[0];
  const float* static_ch_l   = (const float*)d_in[1];
  const float* d_t           = (const float*)d_in[2];
  const float* d_rem         = (const float*)d_in[3];
  const float* dist          = (const float*)d_in[4];
  const float* dec_in        = (const float*)d_in[5];
  const float* h0            = (const float*)d_in[6];
  const float* c0            = (const float*)d_in[7];
  const float* W_ih          = (const float*)d_in[8];
  const float* W_hh          = (const float*)d_in[9];
  const float* Wd            = (const float*)d_in[10];
  const float* bd            = (const float*)d_in[11];
  const float* Wdr           = (const float*)d_in[12];
  const float* bdr           = (const float*)d_in[13];
  const float* Wdist         = (const float*)d_in[14];
  const float* bdist         = (const float*)d_in[15];
  const float* Wchl          = (const float*)d_in[16];
  const float* bchl          = (const float*)d_in[17];
  const float* Wref          = (const float*)d_in[18];
  const float* bref          = (const float*)d_in[19];
  const float* Wq            = (const float*)d_in[20];
  const float* bq            = (const float*)d_in[21];
  const float* v             = (const float*)d_in[22];

  float* out = (float*)d_out;     // [logits 32000 | h_new 16384 | c_new 16384]
  float* ws  = (float*)d_ws;
  float* gates = ws;                          // 65536 f32
  float* hcopy = ws + 65536;                  // 16384 f32
  float* qplus = ws + 81920;                  // 16384 f32
  _Float16* Wf16 = (_Float16*)(ws + 98304);   // 512*2560 f16 (2.62 MB)

  k_prep_w<<<(512 * KTOT) / 256, 256, 0, stream>>>(Wref, Wd, Wdr, Wdist, Wchl, Wf16);
  k_gates<<<4, 256, 0, stream>>>(dec_in, h0, W_ih, W_hh, gates);
  k_cell<<<64, 256, 0, stream>>>(gates, c0, out, hcopy);
  k_qplus<<<64, 256, 0, stream>>>(hcopy, Wq, bq, bd, bdr, bdist, bchl, bref, qplus);
  // X order matches Wcat segments: Wref->static_hidden, Wd->d, Wdr->d_rem,
  // Wdist->dist, Wchl->static_ch_l
  k_main<<<NCOLS / NTILE, 512, 0, stream>>>(Wf16, static_hidden, d_t, d_rem, dist,
                                            static_ch_l, qplus, v, out);
}
